// DAAA_24481313587848
// MI455X (gfx1250) — matrix operand order repaired and checked by reading, not compiled, not run on hardware
//
#include <hip/hip_runtime.h>
#include <math.h>

// ---------------------------------------------------------------------------
// Types for WMMA (gfx1250, wave32)
// ---------------------------------------------------------------------------
typedef __attribute__((ext_vector_type(16))) __bf16 v16bf;
typedef __attribute__((ext_vector_type(8)))  __bf16 v8bf;
typedef __attribute__((ext_vector_type(8)))  float  v8f;

#define F_DIM 128
#define H_DIM 128
#define HHALF 64

__device__ __forceinline__ float sigmoidf_(float x) {
    return 1.0f / (1.0f + expf(-x));
}

__device__ __forceinline__ __bf16 f2bf(float f) {
    unsigned int u = __float_as_uint(f);
    unsigned int r = (u + 0x7FFFu + ((u >> 16) & 1u)) >> 16;
    unsigned short s = (unsigned short)r;
    return *reinterpret_cast<__bf16*>(&s);
}

__device__ __forceinline__ float wave32_sum(float v) {
    #pragma unroll
    for (int o = 16; o > 0; o >>= 1) v += __shfl_xor(v, o, 32);
    return v;
}

// ---------------------------------------------------------------------------
// 1) Feature-importance gating: xg = x * sigmoid(fi[f]); also bf16 copy
// ---------------------------------------------------------------------------
__global__ void k_gate(const float* __restrict__ x, const float* __restrict__ fi,
                       float* __restrict__ xg, __bf16* __restrict__ xg_bf, int total) {
    int i = blockIdx.x * blockDim.x + threadIdx.x;
    if (i >= total) return;
    int f = i & (F_DIM - 1);
    float v = x[i] * sigmoidf_(fi[f]);
    xg[i] = v;
    xg_bf[i] = f2bf(v);
}

// ---------------------------------------------------------------------------
// 2) Weight transpose + bf16 convert: Wt[j*128 + k] = bf16(W[k*J + j])
//    (column-major so WMMA B fragments are two contiguous 8-element loads)
// ---------------------------------------------------------------------------
__global__ void k_wcvt(const float* __restrict__ W, __bf16* __restrict__ Wt,
                       int J, int total) {
    int i = blockIdx.x * blockDim.x + threadIdx.x;
    if (i >= total) return;
    int k = i / J;
    int j = i - k * J;
    Wt[(size_t)j * F_DIM + k] = f2bf(W[(size_t)k * J + j]);
}

// ---------------------------------------------------------------------------
// 3) Edge aggregation: one wave per edge. neighbor_sum[src] += xg[dst]; deg[src]++
// ---------------------------------------------------------------------------
__global__ __launch_bounds__(256)
void k_edge_agg(const int* __restrict__ ei, const float* __restrict__ xg,
                float* __restrict__ nsum, float* __restrict__ deg, int e) {
    int wid  = (blockIdx.x * blockDim.x + threadIdx.x) >> 5;
    int lane = threadIdx.x & 31;
    if (wid >= e) return;
    int s = ei[wid];
    int d = ei[e + wid];
    size_t sb = (size_t)s * F_DIM + lane * 4;
    size_t db = (size_t)d * F_DIM + lane * 4;
    float4 xv = *(const float4*)(xg + db);
    atomicAdd(&nsum[sb + 0], xv.x);
    atomicAdd(&nsum[sb + 1], xv.y);
    atomicAdd(&nsum[sb + 2], xv.z);
    atomicAdd(&nsum[sb + 3], xv.w);
    if (lane == 0) atomicAdd(&deg[s], 1.0f);
}

// ---------------------------------------------------------------------------
// 4) Per-node: mean, cosine dilution, gate; emit bf16 mean and (x+mean)/2
// ---------------------------------------------------------------------------
__global__ __launch_bounds__(256)
void k_node_stats(const float* __restrict__ xg, const float* __restrict__ nsum,
                  const float* __restrict__ deg,
                  const float* __restrict__ gw, const float* __restrict__ gb,
                  __bf16* __restrict__ mean_bf, __bf16* __restrict__ avg_bf,
                  float* __restrict__ gate, int n) {
    int wid  = (blockIdx.x * blockDim.x + threadIdx.x) >> 5;
    int lane = threadIdx.x & 31;
    if (wid >= n) return;
    size_t base = (size_t)wid * F_DIM + lane * 4;
    float4 xv = *(const float4*)(xg + base);
    float4 sv = *(const float4*)(nsum + base);
    float dg = deg[wid];
    float inv = 1.0f / fmaxf(dg, 1.0f);
    float m0 = sv.x * inv, m1 = sv.y * inv, m2 = sv.z * inv, m3 = sv.w * inv;

    mean_bf[base + 0] = f2bf(m0);  mean_bf[base + 1] = f2bf(m1);
    mean_bf[base + 2] = f2bf(m2);  mean_bf[base + 3] = f2bf(m3);
    avg_bf[base + 0] = f2bf((xv.x + m0) * 0.5f);
    avg_bf[base + 1] = f2bf((xv.y + m1) * 0.5f);
    avg_bf[base + 2] = f2bf((xv.z + m2) * 0.5f);
    avg_bf[base + 3] = f2bf((xv.w + m3) * 0.5f);

    float dxx = xv.x * xv.x + xv.y * xv.y + xv.z * xv.z + xv.w * xv.w;
    float dmm = m0 * m0 + m1 * m1 + m2 * m2 + m3 * m3;
    float dxm = xv.x * m0 + xv.y * m1 + xv.z * m2 + xv.w * m3;
    dxx = wave32_sum(dxx);
    dmm = wave32_sum(dmm);
    dxm = wave32_sum(dxm);

    if (lane == 0) {
        float nx = fmaxf(sqrtf(dxx), 1e-12f);
        float nm = fmaxf(sqrtf(dmm), 1e-12f);
        float sim = dxm / (nx * nm);
        if (!(dg > 0.0f)) sim = 1.0f;
        float delta = sigmoidf_(dg * (1.0f - sim) * 0.1f - 0.5f);
        gate[wid] = sigmoidf_(gw[0] * delta + gb[0]);
    }
}

// ---------------------------------------------------------------------------
// 5) Fused WMMA GEMM: one wave per 16x16 output tile.
//    c1 = avg @ W_mean ; c2 = (xg|mean) @ (W_ego|W_nb) ; gate blend; BN stats.
//    jt in [0,8): column tile, uniform per wave (blockDim=256 -> 8 waves/block)
// ---------------------------------------------------------------------------
__global__ __launch_bounds__(256)
void k_fused_gemm(const __bf16* __restrict__ avg_bf,
                  const __bf16* __restrict__ xg_bf,
                  const __bf16* __restrict__ mean_bf,
                  const __bf16* __restrict__ WtM,
                  const __bf16* __restrict__ WtE,
                  const __bf16* __restrict__ WtN,
                  const float* __restrict__ b_mean,
                  const float* __restrict__ b_ego,
                  const float* __restrict__ b_nb,
                  const float* __restrict__ gate,
                  float* __restrict__ h,
                  float* __restrict__ bnsum,
                  float* __restrict__ bnss) {
    const int lane  = threadIdx.x & 31;
    const int jt    = threadIdx.x >> 5;            // column tile 0..7 (uniform per wave)
    const int row0  = blockIdx.x * 16;             // row tile base
    const int colL  = lane & 15;
    const int col   = jt * 16 + colL;              // output column (0..127)
    const int aHalf = (lane >= 16) ? 8 : 0;        // A-fragment K offset
    const int bHalf = (lane >= 16) ? 8 : 0;        // B-fragment K offset
    const int arow  = row0 + colL;                 // A row held by this lane

    const bool second = (jt >= 4);
    const __bf16* __restrict__ A2base = second ? mean_bf : xg_bf;
    const __bf16* __restrict__ Bc = second ? (WtN + (size_t)(col - HHALF) * F_DIM)
                                           : (WtE + (size_t)col * F_DIM);
    const __bf16* __restrict__ Bm = WtM + (size_t)col * F_DIM;

    v8f c1 = {};
    v8f c2 = {};
    #pragma unroll
    for (int kk = 0; kk < 4; ++kk) {
        const int kb = kk * 32;
        const __bf16* pa1 = avg_bf + (size_t)arow * F_DIM + kb + aHalf;
        v8bf a1lo = *(const v8bf*)(pa1);
        v8bf a1hi = *(const v8bf*)(pa1 + 16);
        v16bf a1 = __builtin_shufflevector(a1lo, a1hi,
                     0, 1, 2, 3, 4, 5, 6, 7, 8, 9, 10, 11, 12, 13, 14, 15);

        const __bf16* pa2 = A2base + (size_t)arow * F_DIM + kb + aHalf;
        v8bf a2lo = *(const v8bf*)(pa2);
        v8bf a2hi = *(const v8bf*)(pa2 + 16);
        v16bf a2 = __builtin_shufflevector(a2lo, a2hi,
                     0, 1, 2, 3, 4, 5, 6, 7, 8, 9, 10, 11, 12, 13, 14, 15);

        v16bf bm = __builtin_shufflevector(*(const v8bf*)(Bm + kb + bHalf), *(const v8bf*)(Bm + kb + bHalf + 16),
                     0, 1, 2, 3, 4, 5, 6, 7, 8, 9, 10, 11, 12, 13, 14, 15);
        v16bf bc = __builtin_shufflevector(*(const v8bf*)(Bc + kb + bHalf), *(const v8bf*)(Bc + kb + bHalf + 16),
                     0, 1, 2, 3, 4, 5, 6, 7, 8, 9, 10, 11, 12, 13, 14, 15);

        c1 = __builtin_amdgcn_wmma_f32_16x16x32_bf16(false, a1, false, bm,
                                                     (short)0, c1, false, false);
        c2 = __builtin_amdgcn_wmma_f32_16x16x32_bf16(false, a2, false, bc,
                                                     (short)0, c2, false, false);
    }

    const float bm_c = b_mean[col];
    const float bc_c = second ? b_nb[col - HHALF] : b_ego[col];
    float s = 0.0f, ss = 0.0f;
    #pragma unroll
    for (int v = 0; v < 8; ++v) {
        int row = row0 + v + aHalf;                // C/D layout: M = v + 8*(lane>=16)
        float g  = gate[row];
        float hv = (1.0f - g) * (c1[v] + bm_c) + g * (c2[v] + bc_c);
        h[(size_t)row * H_DIM + col] = hv;
        s  += hv;
        ss += hv * hv;
    }
    s  += __shfl_xor(s, 16, 32);                   // merge lane pair sharing `col`
    ss += __shfl_xor(ss, 16, 32);
    if (lane < 16) {
        atomicAdd(&bnsum[col], s);
        atomicAdd(&bnss[col], ss);
    }
}

// Scalar tail for rows not covered by full 16-row tiles (n % 16 != 0).
__global__ void k_gemm_tail(const float* __restrict__ xg, const float* __restrict__ nsum,
                            const float* __restrict__ deg,
                            const float* __restrict__ W_mean, const float* __restrict__ b_mean,
                            const float* __restrict__ W_ego,  const float* __restrict__ b_ego,
                            const float* __restrict__ W_nb,   const float* __restrict__ b_nb,
                            const float* __restrict__ gate,
                            float* __restrict__ h, float* __restrict__ bnsum,
                            float* __restrict__ bnss, int rowStart, int n) {
    int idx = blockIdx.x * blockDim.x + threadIdx.x;
    int row = rowStart + idx / H_DIM;
    int col = idx & (H_DIM - 1);
    if (row >= n) return;
    float inv = 1.0f / fmaxf(deg[row], 1.0f);
    float accM = 0.0f, accC = 0.0f;
    for (int k = 0; k < F_DIM; ++k) {
        float xv = xg[(size_t)row * F_DIM + k];
        float mv = nsum[(size_t)row * F_DIM + k] * inv;
        accM += (xv + mv) * 0.5f * W_mean[(size_t)k * H_DIM + col];
        accC += (col < HHALF) ? xv * W_ego[(size_t)k * HHALF + col]
                              : mv * W_nb[(size_t)k * HHALF + (col - HHALF)];
    }
    float g = gate[row];
    float bc = (col < HHALF) ? b_ego[col] : b_nb[col - HHALF];
    float hv = (1.0f - g) * (accM + b_mean[col]) + g * (accC + bc);
    h[(size_t)row * H_DIM + col] = hv;
    atomicAdd(&bnsum[col], hv);
    atomicAdd(&bnss[col], hv * hv);
}

// ---------------------------------------------------------------------------
// 6) BN finalize: alpha = gamma*rsqrt(var+eps); betap = beta - mu*alpha
// ---------------------------------------------------------------------------
__global__ void k_bn_finalize(const float* __restrict__ bnsum, const float* __restrict__ bnss,
                              const float* __restrict__ gamma, const float* __restrict__ beta,
                              float* __restrict__ alpha, float* __restrict__ betap,
                              float invN) {
    int f = threadIdx.x;
    if (f >= H_DIM) return;
    float mu  = bnsum[f] * invN;
    float var = bnss[f] * invN - mu * mu;
    float a = gamma[f] * rsqrtf(var + 1e-5f);
    alpha[f] = a;
    betap[f] = beta[f] - mu * a;
}

// ---------------------------------------------------------------------------
// 7) GCN node: BN+ReLU, hw = h @ W_gcn, dinv = rsqrt(deg+1), self-loop + bias
// ---------------------------------------------------------------------------
__global__ __launch_bounds__(256)
void k_gcn_node(const float* __restrict__ h, const float* __restrict__ alpha,
                const float* __restrict__ betap, const float* __restrict__ Wg,
                const float* __restrict__ bg, const float* __restrict__ deg,
                float* __restrict__ hw, float* __restrict__ dinv,
                float* __restrict__ out, int n) {
    int wid  = (blockIdx.x * blockDim.x + threadIdx.x) >> 5;
    int lane = threadIdx.x & 31;
    if (wid >= n) return;
    float a0 = 0.0f, a1 = 0.0f;
    #pragma unroll
    for (int i = 0; i < 4; ++i) {
        int f = lane * 4 + i;
        float hv = alpha[f] * h[(size_t)wid * H_DIM + f] + betap[f];
        hv = fmaxf(hv, 0.0f);
        a0 += hv * Wg[f * 2 + 0];
        a1 += hv * Wg[f * 2 + 1];
    }
    a0 = wave32_sum(a0);
    a1 = wave32_sum(a1);
    if (lane == 0) {
        float dv = rsqrtf(deg[wid] + 1.0f);
        dinv[wid] = dv;
        hw[wid * 2 + 0] = a0;
        hw[wid * 2 + 1] = a1;
        out[wid * 2 + 0] = dv * dv * a0 + bg[0];
        out[wid * 2 + 1] = dv * dv * a1 + bg[1];
    }
}

// ---------------------------------------------------------------------------
// 8) GCN edge scatter: out[src] += dinv[src]*dinv[dst] * hw[dst]
// ---------------------------------------------------------------------------
__global__ void k_gcn_edge(const int* __restrict__ ei, const float* __restrict__ hw,
                           const float* __restrict__ dinv, float* __restrict__ out, int e) {
    int i = blockIdx.x * blockDim.x + threadIdx.x;
    if (i >= e) return;
    int s = ei[i];
    int d = ei[e + i];
    float c = dinv[s] * dinv[d];
    atomicAdd(&out[s * 2 + 0], c * hw[d * 2 + 0]);
    atomicAdd(&out[s * 2 + 1], c * hw[d * 2 + 1]);
}

// ---------------------------------------------------------------------------
// Host side
// ---------------------------------------------------------------------------
static inline size_t alignUp(size_t x) { return (x + 255) & ~(size_t)255; }

extern "C" void kernel_launch(void* const* d_in, const int* in_sizes, int n_in,
                              void* d_out, int out_size, void* d_ws, size_t ws_size,
                              hipStream_t stream) {
    const float* x       = (const float*)d_in[0];
    const int*   ei      = (const int*)  d_in[1];
    const float* fi      = (const float*)d_in[2];
    const float* W_mean  = (const float*)d_in[3];
    const float* b_mean  = (const float*)d_in[4];
    const float* W_ego   = (const float*)d_in[5];
    const float* b_ego   = (const float*)d_in[6];
    const float* W_nb    = (const float*)d_in[7];
    const float* b_nb    = (const float*)d_in[8];
    const float* gate_w  = (const float*)d_in[9];
    const float* gate_b  = (const float*)d_in[10];
    const float* bn_g    = (const float*)d_in[11];
    const float* bn_b    = (const float*)d_in[12];
    const float* W_gcn   = (const float*)d_in[13];
    const float* b_gcn   = (const float*)d_in[14];

    const int n = in_sizes[0] / F_DIM;     // nodes
    const int e = in_sizes[1] / 2;         // edges
    const size_t nf = (size_t)n * F_DIM;

    // Workspace carve-out
    char* p = (char*)d_ws;
    size_t off = 0;
    float*  xg      = (float*) (p + off); off = alignUp(off + nf * 4);
    __bf16* xg_bf   = (__bf16*)(p + off); off = alignUp(off + nf * 2);
    float*  nsum    = (float*) (p + off); off = alignUp(off + nf * 4);
    __bf16* mean_bf = (__bf16*)(p + off); off = alignUp(off + nf * 2);
    __bf16* avg_bf  = (__bf16*)(p + off); off = alignUp(off + nf * 2);
    float*  deg     = (float*) (p + off); off = alignUp(off + (size_t)n * 4);
    float*  gate    = (float*) (p + off); off = alignUp(off + (size_t)n * 4);
    float*  h       = (float*) (p + off); off = alignUp(off + (size_t)n * H_DIM * 4);
    float*  bnsum   = (float*) (p + off); off = alignUp(off + H_DIM * 4);
    float*  bnss    = (float*) (p + off); off = alignUp(off + H_DIM * 4);
    float*  alpha   = (float*) (p + off); off = alignUp(off + H_DIM * 4);
    float*  betap   = (float*) (p + off); off = alignUp(off + H_DIM * 4);
    float*  hw      = (float*) (p + off); off = alignUp(off + (size_t)n * 2 * 4);
    float*  dinv    = (float*) (p + off); off = alignUp(off + (size_t)n * 4);
    __bf16* WtM     = (__bf16*)(p + off); off = alignUp(off + (size_t)F_DIM * H_DIM * 2);
    __bf16* WtE     = (__bf16*)(p + off); off = alignUp(off + (size_t)F_DIM * HHALF * 2);
    __bf16* WtN     = (__bf16*)(p + off); off = alignUp(off + (size_t)F_DIM * HHALF * 2);
    (void)ws_size; (void)n_in; (void)out_size;

    // Zero accumulation buffers (graph-capturable memset nodes)
    hipMemsetAsync(nsum,  0, nf * 4,          stream);
    hipMemsetAsync(deg,   0, (size_t)n * 4,   stream);
    hipMemsetAsync(bnsum, 0, H_DIM * 4,       stream);
    hipMemsetAsync(bnss,  0, H_DIM * 4,       stream);

    // 1) gate
    k_gate<<<(int)((nf + 255) / 256), 256, 0, stream>>>(x, fi, xg, xg_bf, (int)nf);

    // 2) weight conversions (transpose -> bf16 col-major)
    k_wcvt<<<(F_DIM * H_DIM + 255) / 256, 256, 0, stream>>>(W_mean, WtM, H_DIM, F_DIM * H_DIM);
    k_wcvt<<<(F_DIM * HHALF + 255) / 256, 256, 0, stream>>>(W_ego,  WtE, HHALF, F_DIM * HHALF);
    k_wcvt<<<(F_DIM * HHALF + 255) / 256, 256, 0, stream>>>(W_nb,   WtN, HHALF, F_DIM * HHALF);

    // 3) edge aggregation (one wave per edge; 8 waves / 256-thread block)
    k_edge_agg<<<(e + 7) / 8, 256, 0, stream>>>(ei, xg, nsum, deg, e);

    // 4) node stats / gate
    k_node_stats<<<(n + 7) / 8, 256, 0, stream>>>(xg, nsum, deg, gate_w, gate_b,
                                                  mean_bf, avg_bf, gate, n);

    // 5) fused WMMA GEMM over full 16-row tiles + scalar tail
    int ntile = n / 16;
    if (ntile > 0) {
        k_fused_gemm<<<ntile, 256, 0, stream>>>(avg_bf, xg_bf, mean_bf, WtM, WtE, WtN,
                                                b_mean, b_ego, b_nb, gate, h, bnsum, bnss);
    }
    int tailRows = n - ntile * 16;
    if (tailRows > 0) {
        int tailElems = tailRows * H_DIM;
        k_gemm_tail<<<(tailElems + 255) / 256, 256, 0, stream>>>(
            xg, nsum, deg, W_mean, b_mean, W_ego, b_ego, W_nb, b_nb,
            gate, h, bnsum, bnss, ntile * 16, n);
    }

    // 6) BN finalize
    k_bn_finalize<<<1, 128, 0, stream>>>(bnsum, bnss, bn_g, bn_b, alpha, betap,
                                         1.0f / (float)n);

    // 7) GCN node (also initializes d_out with self-loop + bias)
    k_gcn_node<<<(n + 7) / 8, 256, 0, stream>>>(h, alpha, betap, W_gcn, b_gcn, deg,
                                                hw, dinv, (float*)d_out, n);

    // 8) GCN edge scatter
    k_gcn_edge<<<(e + 255) / 256, 256, 0, stream>>>(ei, hw, dinv, (float*)d_out, e);
}
